// NoisyTopKGating_86809878986950
// MI455X (gfx1250) — matrix operand order repaired and checked by reading, not compiled, not run on hardware
//
#include <hip/hip_runtime.h>
#include <math.h>

typedef __attribute__((ext_vector_type(16))) _Float16 v16h;
typedef __attribute__((ext_vector_type(8)))  _Float16 v8h;
typedef __attribute__((ext_vector_type(8)))  float    v8f;

#define D_IN       960
#define NK1        30          // 960/32 K-steps for GEMM1
#define NT1        9           // 8 n-tiles of w1 (128 cols) + 1 n-tile of wn (16 cols)
#define NK2        4           // 128/32 K-steps for GEMM2
#define NT2        2           // 32/16 n-tiles for GEMM2
#define TILE_H     512         // halves per swizzled 32x16 B tile
#define WS_W2_OFF  (NK1*NT1*TILE_H)             // 270*512
#define WS_W3_OFF  ((NK1*NT1 + NK2*NT2)*TILE_H) // 278*512
#define LDS_STRIDE 136         // 128 + 8 halves pad (row stride), 272B = 16B aligned

__device__ __forceinline__ float gelu_exact(float v) {
    return 0.5f * v * (1.0f + erff(v * 0.70710678118654752f));
}

// ---------------------------------------------------------------------------
// Prep: convert fp32 weights -> f16 in WMMA B-layout tiles.
// B-layout (16-bit, 32x16): lane l holds column n = l&15; element e holds
// K = (l<16 ? 0 : 8) + e + (e>=8 ? 8 : 0)  (matches A-layout segments).
// Tile storage: ws[tile*512 + l*16 + e]  -> one coalesced 32B load per lane.
// ---------------------------------------------------------------------------
__global__ __launch_bounds__(32) void prep_weights_k(
    const float* __restrict__ w1, const float* __restrict__ wn,
    const float* __restrict__ w2, const float* __restrict__ w3,
    _Float16* __restrict__ ws)
{
    const int tile = blockIdx.x;          // 0..278
    const int l    = threadIdx.x;         // 0..31
    const int n    = l & 15;
    const int kb   = (l < 16) ? 0 : 8;
    _Float16* dst = ws + (size_t)tile * TILE_H + l * 16;

    if (tile < NK1 * NT1) {                      // w1 | wn tiles
        const int s = tile / NT1;                // K-step
        const int t = tile % NT1;                // n-tile
        if (t < 8) {
            const float* src = w1 + t * 16 + n;  // [960,128] row-major
            #pragma unroll
            for (int e = 0; e < 16; ++e)
                dst[e] = (_Float16)src[(size_t)(s * 32 + kb + e + (e >= 8 ? 8 : 0)) * 128];
        } else {
            const float* src = wn + n;           // [960,16] row-major
            #pragma unroll
            for (int e = 0; e < 16; ++e)
                dst[e] = (_Float16)src[(size_t)(s * 32 + kb + e + (e >= 8 ? 8 : 0)) * 16];
        }
    } else if (tile < NK1 * NT1 + NK2 * NT2) {   // w2 tiles: [128,32]
        const int local = tile - NK1 * NT1;
        const int s = local >> 1;
        const int t = local & 1;
        #pragma unroll
        for (int e = 0; e < 16; ++e)
            dst[e] = (_Float16)w2[(size_t)(s * 32 + kb + e + (e >= 8 ? 8 : 0)) * 32 + t * 16 + n];
    } else {                                     // w3 tile: [32,16]
        #pragma unroll
        for (int e = 0; e < 16; ++e)
            dst[e] = (_Float16)w3[(size_t)(kb + e + (e >= 8 ? 8 : 0)) * 16 + n];
    }
}

// ---------------------------------------------------------------------------
// Main fused gating kernel. 8 waves/block, TWO 16-row M-tiles per wave
// (register blocking: each B tile load feeds 2 WMMAs -> halves L2 traffic).
// ---------------------------------------------------------------------------
__global__ __launch_bounds__(256) void gating_k(
    const float* __restrict__ x,
    const float* __restrict__ b1, const float* __restrict__ g1, const float* __restrict__ be1,
    const float* __restrict__ b2, const float* __restrict__ g2, const float* __restrict__ be2,
    const float* __restrict__ noise,
    const _Float16* __restrict__ ws,
    float* __restrict__ outw, int* __restrict__ outi, float* __restrict__ outc,
    int Brows)
{
    __shared__ __align__(16) _Float16 lds[8][16 * LDS_STRIDE];

    const int wave = threadIdx.x >> 5;
    const int lane = threadIdx.x & 31;
    const int n    = lane & 15;        // C/B layout column within tile
    const int hi   = lane >> 4;        // half: rows 0..7 (lo) / 8..15 (hi)
    const int baseRow = (blockIdx.x * 8 + wave) * 32;
    if (baseRow + 32 > Brows) return;  // wave-uniform; EXEC stays all-ones

    const int k0off = hi ? 8 : 0;      // A-layout K base offset per lane half
    v8f vzero = {};

    // ---------------- GEMM1: x[32x960] @ [w1|wn] ----------------
    v8f acc[2][NT1];                   // [m-tile][n-tile]
    #pragma unroll
    for (int u = 0; u < 2; ++u)
        #pragma unroll
        for (int t = 0; t < NT1; ++t) acc[u][t] = vzero;

    // A-layout (16-bit, 16x32): lane holds row m = lane&15; segments
    // K = [k0, k0+8) and [k0+16, k0+24), k0 = s*32 + (lane<16?0:8).
    const float* xrow0 = x + (size_t)(baseRow + n) * D_IN;
    const float* xrow1 = xrow0 + (size_t)16 * D_IN;

    for (int s = 0; s < NK1; ++s) {
        const int k0 = s * 32 + k0off;
        v16h a0, a1;
        {
            float4 c0 = *(const float4*)(xrow0 + k0);
            float4 c1 = *(const float4*)(xrow0 + k0 + 4);
            float4 c2 = *(const float4*)(xrow0 + k0 + 16);
            float4 c3 = *(const float4*)(xrow0 + k0 + 20);
            a0[0]=(_Float16)c0.x; a0[1]=(_Float16)c0.y; a0[2]=(_Float16)c0.z; a0[3]=(_Float16)c0.w;
            a0[4]=(_Float16)c1.x; a0[5]=(_Float16)c1.y; a0[6]=(_Float16)c1.z; a0[7]=(_Float16)c1.w;
            a0[8]=(_Float16)c2.x; a0[9]=(_Float16)c2.y; a0[10]=(_Float16)c2.z; a0[11]=(_Float16)c2.w;
            a0[12]=(_Float16)c3.x; a0[13]=(_Float16)c3.y; a0[14]=(_Float16)c3.z; a0[15]=(_Float16)c3.w;
        }
        {
            float4 c0 = *(const float4*)(xrow1 + k0);
            float4 c1 = *(const float4*)(xrow1 + k0 + 4);
            float4 c2 = *(const float4*)(xrow1 + k0 + 16);
            float4 c3 = *(const float4*)(xrow1 + k0 + 20);
            a1[0]=(_Float16)c0.x; a1[1]=(_Float16)c0.y; a1[2]=(_Float16)c0.z; a1[3]=(_Float16)c0.w;
            a1[4]=(_Float16)c1.x; a1[5]=(_Float16)c1.y; a1[6]=(_Float16)c1.z; a1[7]=(_Float16)c1.w;
            a1[8]=(_Float16)c2.x; a1[9]=(_Float16)c2.y; a1[10]=(_Float16)c2.z; a1[11]=(_Float16)c2.w;
            a1[12]=(_Float16)c3.x; a1[13]=(_Float16)c3.y; a1[14]=(_Float16)c3.z; a1[15]=(_Float16)c3.w;
        }

        const _Float16* bt = ws + (size_t)(s * NT1) * TILE_H + lane * 16;
        #pragma unroll
        for (int t = 0; t < NT1; ++t) {
            v16h b = *(const v16h*)(bt + t * TILE_H);     // one 32B load, 2 WMMAs
            acc[0][t] = __builtin_amdgcn_wmma_f32_16x16x32_f16(
                false, a0, false, b, (short)0, acc[0][t], false, false);
            acc[1][t] = __builtin_amdgcn_wmma_f32_16x16x32_f16(
                false, a1, false, b, (short)0, acc[1][t], false, false);
        }
    }

    // ---------------- per-column constants (layers 1 & 2) ----------------
    float g1v[8], be1v[8], b1v[8];
    #pragma unroll
    for (int t = 0; t < 8; ++t) {
        b1v[t]  = b1[t * 16 + n];
        g1v[t]  = g1[t * 16 + n];
        be1v[t] = be1[t * 16 + n];
    }
    const float b2v[2]  = { b2[n],  b2[16 + n]  };
    const float g2v[2]  = { g2[n],  g2[16 + n]  };
    const float be2v[2] = { be2[n], be2[16 + n] };

    _Float16* ldw = &lds[wave][0];

    // ---------------- post-pipeline per 16-row half (reuses LDS bounce) ----
    #pragma unroll
    for (int u = 0; u < 2; ++u) {
        // ---- bias + LN(128) + GELU -> LDS (transposed bounce) ----
        #pragma unroll
        for (int r = 0; r < 8; ++r) {
            float s = 0.f, q = 0.f;
            #pragma unroll
            for (int t = 0; t < 8; ++t) {
                float v = acc[u][t][r] + b1v[t];
                acc[u][t][r] = v;
                s += v; q += v * v;
            }
            #pragma unroll
            for (int m = 1; m < 16; m <<= 1) {
                s += __shfl_xor(s, m, 32);
                q += __shfl_xor(q, m, 32);
            }
            const float mu  = s * (1.0f / 128.0f);
            const float var = q * (1.0f / 128.0f) - mu * mu;
            const float inv = rsqrtf(var + 1e-5f);
            const int rowL = r + hi * 8;   // C layout: vgpr r -> M = r (+8 hi half)
            #pragma unroll
            for (int t = 0; t < 8; ++t) {
                float hv = (acc[u][t][r] - mu) * inv * g1v[t] + be1v[t];
                hv = gelu_exact(hv);
                ldw[rowL * LDS_STRIDE + t * 16 + n] = (_Float16)hv;
            }
        }
        __syncthreads();

        // ---- GEMM2: h1[16x128] @ w2[128x32] ----
        v8f acc2[NT2];
        acc2[0] = vzero; acc2[1] = vzero;
        #pragma unroll
        for (int s = 0; s < NK2; ++s) {
            const int k0 = s * 32 + k0off;
            v8h s1 = *(const v8h*)(ldw + n * LDS_STRIDE + k0);
            v8h s2 = *(const v8h*)(ldw + n * LDS_STRIDE + k0 + 16);
            v16h a;
            #pragma unroll
            for (int e = 0; e < 8; ++e) { a[e] = s1[e]; a[8 + e] = s2[e]; }
            const _Float16* bt = ws + WS_W2_OFF + (size_t)(s * NT2) * TILE_H + lane * 16;
            #pragma unroll
            for (int t = 0; t < NT2; ++t) {
                v16h b = *(const v16h*)(bt + t * TILE_H);
                acc2[t] = __builtin_amdgcn_wmma_f32_16x16x32_f16(
                    false, a, false, b, (short)0, acc2[t], false, false);
            }
        }
        __syncthreads();

        // ---- bias + LN(32) + GELU -> LDS ----
        #pragma unroll
        for (int r = 0; r < 8; ++r) {
            float v0 = acc2[0][r] + b2v[0];
            float v1 = acc2[1][r] + b2v[1];
            float s = v0 + v1;
            float q = v0 * v0 + v1 * v1;
            #pragma unroll
            for (int m = 1; m < 16; m <<= 1) {
                s += __shfl_xor(s, m, 32);
                q += __shfl_xor(q, m, 32);
            }
            const float mu  = s * (1.0f / 32.0f);
            const float var = q * (1.0f / 32.0f) - mu * mu;
            const float inv = rsqrtf(var + 1e-5f);
            const int rowL = r + hi * 8;
            float h0 = gelu_exact((v0 - mu) * inv * g2v[0] + be2v[0]);
            float h1 = gelu_exact((v1 - mu) * inv * g2v[1] + be2v[1]);
            ldw[rowL * LDS_STRIDE + n]      = (_Float16)h0;
            ldw[rowL * LDS_STRIDE + 16 + n] = (_Float16)h1;
        }
        __syncthreads();

        // ---- GEMM3: h2[16x32] @ w3[32x16] ----
        v8f acc3 = vzero;
        {
            const int k0 = k0off;
            v8h s1 = *(const v8h*)(ldw + n * LDS_STRIDE + k0);
            v8h s2 = *(const v8h*)(ldw + n * LDS_STRIDE + k0 + 16);
            v16h a;
            #pragma unroll
            for (int e = 0; e < 8; ++e) { a[e] = s1[e]; a[8 + e] = s2[e]; }
            v16h b = *(const v16h*)(ws + WS_W3_OFF + lane * 16);
            acc3 = __builtin_amdgcn_wmma_f32_16x16x32_f16(
                false, a, false, b, (short)0, acc3, false, false);
        }

        // ---- noise, top-2, softmax, outputs ----
        #pragma unroll
        for (int r = 0; r < 8; ++r) {
            const int row = baseRow + u * 16 + hi * 8 + r;
            const float clean = acc3[r];
            const float nl = acc[u][8][r];                     // x @ wn logits
            const float ns = (nl > 20.0f) ? nl : log1pf(expf(nl));  // softplus
            const float nz = noise[(size_t)row * 16 + n];
            const float y  = clean + ns * nz;                  // NOISE_STDDEV = 1

            // top-1 with lowest-index tie-break
            float v1 = y; int i1 = n;
            #pragma unroll
            for (int m = 1; m < 16; m <<= 1) {
                float ov = __shfl_xor(v1, m, 32);
                int   oi = __shfl_xor(i1, m, 32);
                if (ov > v1 || (ov == v1 && oi < i1)) { v1 = ov; i1 = oi; }
            }
            // top-2: mask winner
            float v2 = (n == i1) ? -INFINITY : y; int i2 = n;
            #pragma unroll
            for (int m = 1; m < 16; m <<= 1) {
                float ov = __shfl_xor(v2, m, 32);
                int   oi = __shfl_xor(i2, m, 32);
                if (ov > v2 || (ov == v2 && oi < i2)) { v2 = ov; i2 = oi; }
            }
            if (n == 0) {
                const float d  = expf(v2 - v1);      // TEMPERATURE = 1
                const float wa = 1.0f / (1.0f + d);
                outw[(size_t)row * 2 + 0] = wa;
                outw[(size_t)row * 2 + 1] = 1.0f - wa;
                outi[(size_t)row * 2 + 0] = i1;
                outi[(size_t)row * 2 + 1] = i2;
            }
            outc[(size_t)row * 16 + n] = clean;
        }
        __syncthreads();   // protect LDS bounce region before next half reuses it
    }
}

// ---------------------------------------------------------------------------
extern "C" void kernel_launch(void* const* d_in, const int* in_sizes, int n_in,
                              void* d_out, int out_size, void* d_ws, size_t ws_size,
                              hipStream_t stream) {
    const float* x   = (const float*)d_in[0];
    const float* w1  = (const float*)d_in[1];
    const float* b1  = (const float*)d_in[2];
    const float* g1  = (const float*)d_in[3];
    const float* be1 = (const float*)d_in[4];
    const float* w2  = (const float*)d_in[5];
    const float* b2  = (const float*)d_in[6];
    const float* g2  = (const float*)d_in[7];
    const float* be2 = (const float*)d_in[8];
    const float* w3  = (const float*)d_in[9];
    const float* wn  = (const float*)d_in[10];
    const float* noise = (const float*)d_in[11];

    const int Brows = in_sizes[0] / D_IN;     // 131072

    _Float16* ws = (_Float16*)d_ws;           // needs ~286 KB (swizzled f16 weights)

    float* out  = (float*)d_out;
    float* outw = out;                              // [B,2] combined_weights (f32)
    int*   outi = (int*)(out + (size_t)2 * Brows);  // [B,2] top_k_indices (i32 bits)
    float* outc = out + (size_t)4 * Brows;          // [B,16] clean_logits (f32)

    const int n_tiles = NK1 * NT1 + NK2 * NT2 + 1;  // 279
    prep_weights_k<<<n_tiles, 32, 0, stream>>>(w1, wn, w2, w3, ws);

    const int blocks = (Brows + 255) / 256;    // 8 waves x 32 rows per block
    gating_k<<<blocks, 256, 0, stream>>>(x, b1, g1, be1, b2, g2, be2,
                                         noise, ws, outw, outi, outc, Brows);
}
